// TernaryTransformer_19275813224784
// MI455X (gfx1250) — matrix operand order repaired and checked by reading, not compiled, not run on hardware
//
#include <hip/hip_runtime.h>
#include <hip/hip_bf16.h>
#include <math.h>
#include <stdint.h>

// ---------------- model dims ----------------
#define V_  32000
#define D_  1024
#define H_  16
#define L_  2
#define B_  32
#define S_  100
#define DH_ 64
#define FF_ 4096
#define M_  (B_ * S_)        // 3200 token rows
#define LN_EPS 1e-5f
#define Q_EPS  1e-6f

typedef __bf16 bf16x16 __attribute__((ext_vector_type(16)));
typedef float  floatx8 __attribute__((ext_vector_type(8)));

// ---------------- helpers ----------------
__device__ __forceinline__ unsigned short f32_to_bf16_rne(float f) {
    unsigned int u = __float_as_uint(f);
    u += 0x7FFFu + ((u >> 16) & 1u);   // round to nearest even
    return (unsigned short)(u >> 16);
}

// LDS offset of a __shared__ pointer: flat LDS aperture maps addr[31:0] -> LDS offset
__device__ __forceinline__ unsigned lds_off(const void* p) {
    return (unsigned)(uintptr_t)p;
}

// async global -> LDS, 16 bytes per lane (gfx1250 GLOBAL_LOAD_ASYNC_TO_LDS_B128)
__device__ __forceinline__ void async_load_b128(unsigned dst_lds, const void* src) {
    asm volatile("global_load_async_to_lds_b128 %0, %1, off"
                 :: "v"(dst_lds), "v"((unsigned long long)src) : "memory");
}
__device__ __forceinline__ void wait_async_le3() {   // previous group landed
    asm volatile("s_wait_asynccnt 0x3" ::: "memory");
}
__device__ __forceinline__ void wait_async0() {
    asm volatile("s_wait_asynccnt 0x0" ::: "memory");
}

// ---------------- kernels ----------------

// h[row, :] = embed[x[row], :] + pos_emb[row % S, :]
__global__ __launch_bounds__(256) void k_embed(const int* __restrict__ x,
                                               const float* __restrict__ embed,
                                               const float* __restrict__ pos,
                                               float* __restrict__ h) {
    int row = blockIdx.x;              // 0..M-1
    int s   = row % S_;
    int tok = x[row];
    int d0  = threadIdx.x * 4;
    const float4 e = *reinterpret_cast<const float4*>(embed + (size_t)tok * D_ + d0);
    const float4 p = *reinterpret_cast<const float4*>(pos + (size_t)s * D_ + d0);
    float4 o; o.x = e.x + p.x; o.y = e.y + p.y; o.z = e.z + p.z; o.w = e.w + p.w;
    *reinterpret_cast<float4*>(h + (size_t)row * D_ + d0) = o;
}

// f32 -> bf16 converter (grid-stride)
__global__ __launch_bounds__(256) void k_cvt_bf16(const float* __restrict__ in,
                                                  unsigned short* __restrict__ out,
                                                  long long n) {
    long long i = (long long)blockIdx.x * blockDim.x + threadIdx.x;
    long long stride = (long long)gridDim.x * blockDim.x;
    for (; i < n; i += stride) out[i] = f32_to_bf16_rne(in[i]);
}

// Tiled WMMA GEMM:  C[M,N] (f32) = A[M,K](bf16) * W[N,K]^T(bf16) + bias[N]
// block = 256 threads (8 waves), block tile 128x64, wave tile 32x32, K-step 32.
// Double-buffered LDS: tile t+1 is DMA'd (GLOBAL_LOAD_ASYNC_TO_LDS_B128, ASYNCcnt)
// while tile t feeds 4x v_wmma_f32_16x16x32_bf16 per wave; pipelined wait is
// s_wait_asynccnt 3 (async loads complete in order).
#define LDT 40   // LDS row stride in ushorts (80 B -> conflict-free, keeps 16B align)
__global__ __launch_bounds__(256) void k_gemm_bf16(const unsigned short* __restrict__ A,
                                                   const unsigned short* __restrict__ W,
                                                   const float* __restrict__ bias,
                                                   float* __restrict__ C,
                                                   int M, int N, int K) {
    __shared__ __align__(16) unsigned short sA[2][128 * LDT];   // 2 x 10 KB
    __shared__ __align__(16) unsigned short sB[2][64 * LDT];    // 2 x  5 KB

    const int tid  = threadIdx.x;
    const int lane = tid & 31;
    const int wave = tid >> 5;
    const int mTile = blockIdx.y * 128;
    const int nTile = blockIdx.x * 64;
    const int mw = (wave >> 1) * 32;   // 0,32,64,96
    const int nw = (wave & 1) * 32;    // 0,32

    floatx8 c00 = {}, c01 = {}, c10 = {}, c11 = {};

    // cooperative tile loaders: thread t -> row t/4, 8-bf16 chunk (t%4)*8
    const int lr = tid >> 2;
    const int lc = (tid & 3) * 8;
    const unsigned short* gA0 = A + (size_t)(mTile + lr) * K + lc;        // rows 0..63
    const unsigned short* gA1 = A + (size_t)(mTile + 64 + lr) * K + lc;   // rows 64..127
    const unsigned short* gB0 = W + (size_t)(nTile + lr) * K + lc;        // rows 0..63
    const unsigned sa0[2] = { lds_off(&sA[0][lr * LDT + lc]),       lds_off(&sA[1][lr * LDT + lc]) };
    const unsigned sa1[2] = { lds_off(&sA[0][(64 + lr) * LDT + lc]), lds_off(&sA[1][(64 + lr) * LDT + lc]) };
    const unsigned sb0[2] = { lds_off(&sB[0][lr * LDT + lc]),       lds_off(&sB[1][lr * LDT + lc]) };

    // fragment gather offsets (A: m = lane%16, K chunks at h8 and 16+h8;
    //  B: n = lane%16, K chunks at h8 and 16+h8)
    const int h8  = (lane >> 4) * 8;
    const int am0 = (mw + (lane & 15)) * LDT;
    const int am1 = am0 + 16 * LDT;
    const int bn0 = (nw + (lane & 15)) * LDT;
    const int bn1 = bn0 + 16 * LDT;
    const int kb  = (lane >> 4) * 16;

    const int nsteps = K >> 5;

    // prologue: DMA tile 0 into buffer 0
    async_load_b128(sa0[0], gA0);
    async_load_b128(sa1[0], gA1);
    async_load_b128(sb0[0], gB0);
    gA0 += 32; gA1 += 32; gB0 += 32;

    for (int it = 0; it < nsteps; ++it) {
        const int cur = it & 1;
        const bool more = (it + 1 < nsteps);
        if (more) {                            // DMA tile it+1 into alternate buffer
            const int nxt = cur ^ 1;
            async_load_b128(sa0[nxt], gA0);
            async_load_b128(sa1[nxt], gA1);
            async_load_b128(sb0[nxt], gB0);
            gA0 += 32; gA1 += 32; gB0 += 32;
            __builtin_prefetch((const void*)gA0);     // gfx1250 global_prefetch
            __builtin_prefetch((const void*)gB0);
            wait_async_le3();                  // tile it has landed (in-order)
        } else {
            wait_async0();
        }
        __syncthreads();                       // tile it visible to all waves

        const unsigned short* cA = sA[cur];
        const unsigned short* cB = sB[cur];
        bf16x16 a0, a1, b0, b1;
        {
            uint4* d = reinterpret_cast<uint4*>(&a0);
            d[0] = *reinterpret_cast<const uint4*>(&cA[am0 + h8]);
            d[1] = *reinterpret_cast<const uint4*>(&cA[am0 + 16 + h8]);
        }
        {
            uint4* d = reinterpret_cast<uint4*>(&a1);
            d[0] = *reinterpret_cast<const uint4*>(&cA[am1 + h8]);
            d[1] = *reinterpret_cast<const uint4*>(&cA[am1 + 16 + h8]);
        }
        {
            uint4* d = reinterpret_cast<uint4*>(&b0);
            d[0] = *reinterpret_cast<const uint4*>(&cB[bn0 + h8]);
            d[1] = *reinterpret_cast<const uint4*>(&cB[bn0 + 16 + h8]);
        }
        {
            uint4* d = reinterpret_cast<uint4*>(&b1);
            d[0] = *reinterpret_cast<const uint4*>(&cB[bn1 + h8]);
            d[1] = *reinterpret_cast<const uint4*>(&cB[bn1 + 16 + h8]);
        }
        __syncthreads();                       // frag reads done; buffer reusable

        c00 = __builtin_amdgcn_wmma_f32_16x16x32_bf16(false, a0, false, b0,
                                                      (short)0, c00, false, false);
        c01 = __builtin_amdgcn_wmma_f32_16x16x32_bf16(false, a0, false, b1,
                                                      (short)0, c01, false, false);
        c10 = __builtin_amdgcn_wmma_f32_16x16x32_bf16(false, a1, false, b0,
                                                      (short)0, c10, false, false);
        c11 = __builtin_amdgcn_wmma_f32_16x16x32_bf16(false, a1, false, b1,
                                                      (short)0, c11, false, false);
    }

    // epilogue: C/D layout -> VGPR i holds M = i + (lane>>4)*8, N = lane&15
    const int col0 = nTile + nw + (lane & 15);
    const int col1 = col0 + 16;
    const float bv0 = bias[col0];
    const float bv1 = bias[col1];
    const int r0 = mTile + mw + ((lane >> 4) * 8);
    const int r1 = r0 + 16;
    #pragma unroll
    for (int i = 0; i < 8; ++i) {
        C[(size_t)(r0 + i) * N + col0] = c00[i] + bv0;
        C[(size_t)(r0 + i) * N + col1] = c01[i] + bv1;
        C[(size_t)(r1 + i) * N + col0] = c10[i] + bv0;
        C[(size_t)(r1 + i) * N + col1] = c11[i] + bv1;
    }
}

// attention for one (batch, head): scores -> softmax -> ctx
// qkv: [M, 3D] rows = b*S+s, q at [h*64..], k at [1024+h*64..], v at [2048+h*64..]
__global__ __launch_bounds__(256) void k_attn(const float* __restrict__ qkv,
                                              float* __restrict__ ctx) {
    __shared__ float sc[S_ * S_];              // 40 KB
    const int bh = blockIdx.x;
    const int b  = bh / H_;
    const int hh = bh % H_;
    const int tid = threadIdx.x;

    // scores
    for (int idx = tid; idx < S_ * S_; idx += 256) {
        int i = idx / S_, j = idx % S_;
        const float* qr = qkv + (size_t)(b * S_ + i) * (3 * D_) + hh * DH_;
        const float* kr = qkv + (size_t)(b * S_ + j) * (3 * D_) + D_ + hh * DH_;
        float acc = 0.f;
        #pragma unroll 8
        for (int d = 0; d < DH_; ++d) acc += qr[d] * kr[d];
        sc[idx] = acc * 0.125f;                 // 1/sqrt(64)
    }
    __syncthreads();

    // softmax per row (one thread per row)
    if (tid < S_) {
        float* row = sc + tid * S_;
        float mx = row[0];
        for (int j = 1; j < S_; ++j) mx = fmaxf(mx, row[j]);
        float sum = 0.f;
        for (int j = 0; j < S_; ++j) { float e = expf(row[j] - mx); row[j] = e; sum += e; }
        float inv = 1.f / sum;
        for (int j = 0; j < S_; ++j) row[j] *= inv;
    }
    __syncthreads();

    // ctx[i, d] = sum_j attn[i,j] * v[j,d]
    for (int idx = tid; idx < S_ * DH_; idx += 256) {
        int i = idx / DH_, d = idx % DH_;
        const float* vcol = qkv + (size_t)(b * S_) * (3 * D_) + 2 * D_ + hh * DH_ + d;
        float acc = 0.f;
        const float* arow = sc + i * S_;
        for (int j = 0; j < S_; ++j) acc += arow[j] * vcol[(size_t)j * (3 * D_)];
        ctx[(size_t)(b * S_ + i) * D_ + hh * DH_ + d] = acc;
    }
}

// out[row,:] = LayerNorm(resid[row,:] + add[row,:]) * w + b   (block per row)
__global__ __launch_bounds__(256) void k_add_ln(const float* __restrict__ resid,
                                                const float* __restrict__ add,
                                                const float* __restrict__ w,
                                                const float* __restrict__ bias,
                                                float* __restrict__ out) {
    __shared__ float s1[256];
    __shared__ float s2[256];
    const int row = blockIdx.x;
    const int tid = threadIdx.x;
    float v[4];
    float sum = 0.f, sq = 0.f;
    #pragma unroll
    for (int i = 0; i < 4; ++i) {
        int d = tid + i * 256;
        float t = resid[(size_t)row * D_ + d] + add[(size_t)row * D_ + d];
        v[i] = t; sum += t; sq += t * t;
    }
    s1[tid] = sum; s2[tid] = sq;
    __syncthreads();
    for (int off = 128; off > 0; off >>= 1) {
        if (tid < off) { s1[tid] += s1[tid + off]; s2[tid] += s2[tid + off]; }
        __syncthreads();
    }
    const float mu  = s1[0] * (1.f / D_);
    const float var = s2[0] * (1.f / D_) - mu * mu;
    const float inv = rsqrtf(var + LN_EPS);
    #pragma unroll
    for (int i = 0; i < 4; ++i) {
        int d = tid + i * 256;
        out[(size_t)row * D_ + d] = (v[i] - mu) * inv * w[d] + bias[d];
    }
}

// exact gelu in place
__global__ __launch_bounds__(256) void k_gelu(float* __restrict__ x, long long n) {
    long long i = (long long)blockIdx.x * blockDim.x + threadIdx.x;
    long long stride = (long long)gridDim.x * blockDim.x;
    for (; i < n; i += stride) {
        float t = x[i];
        x[i] = 0.5f * t * (1.f + erff(t * 0.70710678118654752f));
    }
}

// deterministic 2-stage sum of |dec_w|
__global__ __launch_bounds__(256) void k_absmean1(const float* __restrict__ w,
                                                  float* __restrict__ partial,
                                                  long long n) {
    __shared__ float s[256];
    long long i = (long long)blockIdx.x * blockDim.x + threadIdx.x;
    long long stride = (long long)gridDim.x * blockDim.x;
    float acc = 0.f;
    for (; i < n; i += stride) acc += fabsf(w[i]);
    s[threadIdx.x] = acc;
    __syncthreads();
    for (int off = 128; off > 0; off >>= 1) {
        if (threadIdx.x < off) s[threadIdx.x] += s[threadIdx.x + off];
        __syncthreads();
    }
    if (threadIdx.x == 0) partial[blockIdx.x] = s[0];
}

__global__ __launch_bounds__(256) void k_absmean2(const float* __restrict__ partial,
                                                  float* __restrict__ gsum, int np) {
    __shared__ float s[256];
    float acc = 0.f;
    for (int i = threadIdx.x; i < np; i += 256) acc += partial[i];
    s[threadIdx.x] = acc;
    __syncthreads();
    for (int off = 128; off > 0; off >>= 1) {
        if (threadIdx.x < off) s[threadIdx.x] += s[threadIdx.x + off];
        __syncthreads();
    }
    if (threadIdx.x == 0) gsum[0] = s[0];
}

// tw = clip(rint(w / (mean|w| + eps)), -1, 1) as bf16 (exact in bf16)
__global__ __launch_bounds__(256) void k_ternarize(const float* __restrict__ w,
                                                   const float* __restrict__ gsum,
                                                   unsigned short* __restrict__ tw,
                                                   long long n) {
    const float g = gsum[0] / (float)n;
    const float inv = 1.f / (g + Q_EPS);
    long long i = (long long)blockIdx.x * blockDim.x + threadIdx.x;
    long long stride = (long long)gridDim.x * blockDim.x;
    for (; i < n; i += stride) {
        float q = rintf(w[i] * inv);
        q = fminf(1.f, fmaxf(-1.f, q));
        tw[i] = f32_to_bf16_rne(q);
    }
}

// ---------------- launcher ----------------
static inline size_t align256(size_t x) { return (x + 255) & ~(size_t)255; }

extern "C" void kernel_launch(void* const* d_in, const int* in_sizes, int n_in,
                              void* d_out, int out_size, void* d_ws, size_t ws_size,
                              hipStream_t stream) {
    const int*   x      = (const int*)  d_in[0];
    const float* embed  = (const float*)d_in[1];
    const float* pos    = (const float*)d_in[2];
    const float* Wqkv   = (const float*)d_in[3];
    const float* bqkv   = (const float*)d_in[4];
    const float* Wo     = (const float*)d_in[5];
    const float* bo     = (const float*)d_in[6];
    const float* ln1w   = (const float*)d_in[7];
    const float* ln1b   = (const float*)d_in[8];
    const float* W1     = (const float*)d_in[9];
    const float* b1     = (const float*)d_in[10];
    const float* W2     = (const float*)d_in[11];
    const float* b2     = (const float*)d_in[12];
    const float* ln2w   = (const float*)d_in[13];
    const float* ln2b   = (const float*)d_in[14];
    const float* dec_w  = (const float*)d_in[15];
    const float* dec_b  = (const float*)d_in[16];
    float* out = (float*)d_out;

    // workspace carve-up
    char* ws = (char*)d_ws;
    size_t off = 0;
    float* h       = (float*)(ws + off); off = align256(off + (size_t)M_ * D_  * 4);
    float* tmp_big = (float*)(ws + off); off = align256(off + (size_t)M_ * FF_ * 4);
    float* tmp_d   = (float*)(ws + off); off = align256(off + (size_t)M_ * D_  * 4);
    unsigned short* abf = (unsigned short*)(ws + off); off = align256(off + (size_t)M_ * FF_ * 2);
    unsigned short* wbf = (unsigned short*)(ws + off); off = align256(off + (size_t)V_ * D_  * 2);
    float* partial = (float*)(ws + off); off = align256(off + 4096 * 4);
    float* gsum    = (float*)(ws + off); off = align256(off + 4);
    (void)ws_size; (void)n_in; (void)in_sizes; (void)out_size;

    const dim3 blk(256);

    auto cvt = [&](const float* src, unsigned short* dst, long long n) {
        int nb = (int)((n + 256 * 8 - 1) / (256 * 8));
        if (nb > 4096) nb = 4096;
        k_cvt_bf16<<<nb, blk, 0, stream>>>(src, dst, n);
    };
    auto gemm = [&](const unsigned short* A, const unsigned short* W,
                    const float* bias, float* C, int M, int N, int K) {
        dim3 grid(N / 64, M / 128);
        k_gemm_bf16<<<grid, blk, 0, stream>>>(A, W, bias, C, M, N, K);
    };

    // embedding + positional
    k_embed<<<M_, blk, 0, stream>>>(x, embed, pos, h);

    for (int l = 0; l < L_; ++l) {
        const float* Wqkv_l = Wqkv + (size_t)l * 3 * D_ * D_;
        const float* bqkv_l = bqkv + (size_t)l * 3 * D_;
        const float* Wo_l   = Wo   + (size_t)l * D_ * D_;
        const float* bo_l   = bo   + (size_t)l * D_;
        const float* W1_l   = W1   + (size_t)l * FF_ * D_;
        const float* b1_l   = b1   + (size_t)l * FF_;
        const float* W2_l   = W2   + (size_t)l * D_ * FF_;
        const float* b2_l   = b2   + (size_t)l * D_;

        // qkv = h @ Wqkv^T + bqkv
        cvt(h, abf, (long long)M_ * D_);
        cvt(Wqkv_l, wbf, (long long)3 * D_ * D_);
        gemm(abf, wbf, bqkv_l, tmp_big, M_, 3 * D_, D_);

        // attention
        k_attn<<<B_ * H_, blk, 0, stream>>>(tmp_big, tmp_d);

        // attn_out = ctx @ Wo^T + bo
        cvt(tmp_d, abf, (long long)M_ * D_);
        cvt(Wo_l, wbf, (long long)D_ * D_);
        gemm(abf, wbf, bo_l, tmp_big, M_, D_, D_);

        // h = LN(h + attn_out)
        k_add_ln<<<M_, blk, 0, stream>>>(h, tmp_big, ln1w + (size_t)l * D_,
                                         ln1b + (size_t)l * D_, h);

        // ff = gelu(h @ W1^T + b1)
        cvt(h, abf, (long long)M_ * D_);
        cvt(W1_l, wbf, (long long)FF_ * D_);
        gemm(abf, wbf, b1_l, tmp_big, M_, FF_, D_);
        k_gelu<<<4096, blk, 0, stream>>>(tmp_big, (long long)M_ * FF_);

        // ff2 = ff @ W2^T + b2
        cvt(tmp_big, abf, (long long)M_ * FF_);
        cvt(W2_l, wbf, (long long)D_ * FF_);
        gemm(abf, wbf, b2_l, tmp_d, M_, D_, FF_);

        // h = LN(h + ff2)
        k_add_ln<<<M_, blk, 0, stream>>>(h, tmp_d, ln2w + (size_t)l * D_,
                                         ln2b + (size_t)l * D_, h);
    }

    // ternary decoder
    const long long nw = (long long)V_ * D_;
    k_absmean1<<<4096, blk, 0, stream>>>(dec_w, partial, nw);
    k_absmean2<<<1, blk, 0, stream>>>(partial, gsum, 4096);
    k_ternarize<<<4096, blk, 0, stream>>>(dec_w, gsum, wbf, nw);
    cvt(h, abf, (long long)M_ * D_);
    gemm(abf, wbf, dec_b, out, M_, V_, D_);   // logits -> d_out [M, V]
}
